// SparseMoELanguageModel_42202348651207
// MI455X (gfx1250) — matrix operand order repaired and checked by reading, not compiled, not run on hardware
//
#include <hip/hip_runtime.h>

// ---------------- constants (match reference) ----------------
#define Lc    4
#define Ec    512
#define Hc    8      // heads
#define Dhc   64
#define Xc    8      // experts
#define Kc    2      // top-k
#define Bc    8
#define Tc    1024
#define BTc   (Bc*Tc)        // 8192
#define CAPc  (BTc*Kc/Xc)    // 2048
#define HIGHc 32000
#define HIDc  1024

typedef __attribute__((ext_vector_type(16))) _Float16 v16h;
typedef __attribute__((ext_vector_type(8)))  _Float16 v8h;
typedef __attribute__((ext_vector_type(4)))  _Float16 v4h;
typedef __attribute__((ext_vector_type(8)))  float    v8f;

__device__ __forceinline__ v8f wmma_f16(v16h a, v16h b, v8f c) {
  return __builtin_amdgcn_wmma_f32_16x16x32_f16(false, a, false, b, (short)0, c, false, false);
}
__device__ __forceinline__ v16h cat8(v8h lo, v8h hi) {
  return __builtin_shufflevector(lo, hi, 0,1,2,3,4,5,6,7,8,9,10,11,12,13,14,15);
}
__device__ __forceinline__ void cvt4(v16h& d, int base, float4 f) {
  d[base+0] = (_Float16)f.x; d[base+1] = (_Float16)f.y;
  d[base+2] = (_Float16)f.z; d[base+3] = (_Float16)f.w;
}

// ================= generic GEMM: C[M,N] = A[M,K] * W[N,K]^T =================
// Block tile 64(M) x 128(N), k-step 64, LDS ping-pong; 8 waves x 64x16 strip.
// Out-of-range rows/cols are CLAMPED on load (compute garbage, never stored).
// flags: 1=bias, 2=relu, 4=accumulate into C.  Kd must be a multiple of 64.
__global__ __launch_bounds__(256)
void gemm_nt_kernel(const float* __restrict__ A, const float* __restrict__ W,
                    const float* __restrict__ bias, float* __restrict__ C,
                    int M, int N, int Kd, int flags)
{
  __shared__ __align__(16) _Float16 Ash[2][64 * 64];   // 2 x 8 KB ping-pong
  const int tid   = threadIdx.x;
  const int lane  = tid & 31;
  const int wave  = tid >> 5;
  const int g     = lane >> 4;
  const int r     = lane & 15;
  const int m0    = blockIdx.y * 64;
  const int n0    = blockIdx.x * 128 + wave * 16;
  const int nrow  = n0 + r;

  // staging geometry: thread stages 4 float4-groups (one per 16-row band)
  int lrow[4], lkk[4];
  const float* asrc[4];
#pragma unroll
  for (int it = 0; it < 4; ++it) {
    int gidx = tid + it * 256;            // 0..1023
    lrow[it] = gidx >> 4;                 // 0..63
    lkk[it]  = (gidx & 15) << 2;          // 0..60
    int srow = m0 + lrow[it]; if (srow > M - 1) srow = M - 1;   // clamp, branch-free
    asrc[it] = A + (size_t)srow * Kd + lkk[it];
  }
  const int nclamp = (nrow < N) ? nrow : (N - 1);
  const float* wbase = W + (size_t)nclamp * Kd + 16 * g;

  float4 rg[4];
#pragma unroll
  for (int it = 0; it < 4; ++it) rg[it] = *(const float4*)(asrc[it]);

  v8f acc[4] = {{}, {}, {}, {}};
  int buf = 0;

  for (int k0 = 0; k0 < Kd; k0 += 64) {
    // ---- commit staged regs to LDS (f16) ----
#pragma unroll
    for (int it = 0; it < 4; ++it) {
      v4h hv; hv[0] = (_Float16)rg[it].x; hv[1] = (_Float16)rg[it].y;
              hv[2] = (_Float16)rg[it].z; hv[3] = (_Float16)rg[it].w;
      *(v4h*)(&Ash[buf][lrow[it] * 64 + lkk[it]]) = hv;
    }
    __syncthreads();

    // ---- prefetch next A tile into registers (overlaps with compute) ----
    if (k0 + 64 < Kd) {
#pragma unroll
      for (int it = 0; it < 4; ++it) rg[it] = *(const float4*)(asrc[it] + k0 + 64);
    }

    // ---- B fragments: 8 unconditional global_load_b128 (clamped row) ----
    v16h b0, b1;
    {
      const float* wr = wbase + k0;
      cvt4(b0, 0,  *(const float4*)(wr + 0));   cvt4(b0, 4,  *(const float4*)(wr + 4));
      cvt4(b0, 8,  *(const float4*)(wr + 8));   cvt4(b0, 12, *(const float4*)(wr + 12));
      cvt4(b1, 0,  *(const float4*)(wr + 32));  cvt4(b1, 4,  *(const float4*)(wr + 36));
      cvt4(b1, 8,  *(const float4*)(wr + 40));  cvt4(b1, 12, *(const float4*)(wr + 44));
    }

    // ---- batch all 16 ds_load_b128, then 8 WMMAs (single DS drain) ----
    v16h af[4][2];
#pragma unroll
    for (int mt = 0; mt < 4; ++mt) {
      const _Float16* arow = &Ash[buf][(mt * 16 + r) * 64];
      af[mt][0] = cat8(*(const v8h*)(arow +      16 * g), *(const v8h*)(arow + 8 + 16 * g));
      af[mt][1] = cat8(*(const v8h*)(arow + 32 + 16 * g), *(const v8h*)(arow + 40 + 16 * g));
    }
#pragma unroll
    for (int mt = 0; mt < 4; ++mt) {
      acc[mt] = wmma_f16(af[mt][0], b0, acc[mt]);
      acc[mt] = wmma_f16(af[mt][1], b1, acc[mt]);
    }
    buf ^= 1;
  }

  const bool nok = (nrow < N);
#pragma unroll
  for (int mt = 0; mt < 4; ++mt) {
#pragma unroll
    for (int i = 0; i < 8; ++i) {
      int m = m0 + mt * 16 + i + 8 * g;
      if (nok && m < M) {
        float v = acc[mt][i];
        if (flags & 1) v += bias[nrow];
        if (flags & 2) v = fmaxf(v, 0.f);
        if (flags & 4) C[(size_t)m * N + nrow] += v;
        else           C[(size_t)m * N + nrow]  = v;
      }
    }
  }
}

// ================= embedding =================
__global__ void embed_kernel(const float* __restrict__ te, const float* __restrict__ pe,
                             const int* __restrict__ ids, float* __restrict__ x)
{
  int id = blockIdx.x * 256 + threadIdx.x;       // BT*E threads
  int e  = id & (Ec - 1);
  int bt = id >> 9;
  int t  = bt & (Tc - 1);
  x[id] = te[(size_t)ids[bt] * Ec + e] + pe[(size_t)t * Ec + e];
}

// ================= layernorm over E=512 (one row / block of 256) =================
__global__ __launch_bounds__(256)
void layernorm_kernel(const float* __restrict__ in, const float* __restrict__ w,
                      const float* __restrict__ bb, float* __restrict__ out)
{
  __shared__ float red[256];
  const int row = blockIdx.x, tid = threadIdx.x;
  const float* xr = in + (size_t)row * Ec;
  float v0 = xr[tid], v1 = xr[tid + 256];
  red[tid] = v0 + v1; __syncthreads();
  for (int off = 128; off > 0; off >>= 1) { if (tid < off) red[tid] += red[tid + off]; __syncthreads(); }
  float m = red[0] * (1.f / Ec);
  __syncthreads();
  float d0 = v0 - m, d1 = v1 - m;
  red[tid] = d0*d0 + d1*d1; __syncthreads();
  for (int off = 128; off > 0; off >>= 1) { if (tid < off) red[tid] += red[tid + off]; __syncthreads(); }
  float rstd = rsqrtf(red[0] * (1.f / Ec) + 1e-5f);
  float* orow = out + (size_t)row * Ec;
  orow[tid]       = d0 * rstd * w[tid]       + bb[tid];
  orow[tid + 256] = d1 * rstd * w[tid + 256] + bb[tid + 256];
}

// ================= RoPE on q,k halves of qkv buffer =================
__global__ void rope_kernel(float* __restrict__ qkv)
{
  int id = blockIdx.x * 256 + threadIdx.x;       // BT*Hn*HALF
  if (id >= BTc * Hc * 32) return;
  int i  = id & 31;
  int h  = (id >> 5) & 7;
  int bt = id >> 8;
  int t  = bt & (Tc - 1);
  float ang = (float)t * __expf((float)(2 * i) * (-9.210340371976184f / 64.f));
  float sn, cs; __sincosf(ang, &sn, &cs);
  float* qp = qkv + (size_t)bt * (3 * Ec) + h * Dhc;
  float a1 = qp[i], a2 = qp[i + 32];
  qp[i] = a1 * cs - a2 * sn; qp[i + 32] = a2 * cs + a1 * sn;
  float* kp = qp + Ec;
  a1 = kp[i]; a2 = kp[i + 32];
  kp[i] = a1 * cs - a2 * sn; kp[i + 32] = a2 * cs + a1 * sn;
}

// ================= flash attention, WMMA, one wave per (b,h,16-query block) ====
__global__ __launch_bounds__(32)
void attn_kernel(const float* __restrict__ qkv, float* __restrict__ att)
{
  __shared__ __align__(16) _Float16 plds[16][32];
  const int lane = threadIdx.x;
  const int g = lane >> 4, r = lane & 15;
  const int qb = blockIdx.x, h = blockIdx.y, b = blockIdx.z;
  const float* base = qkv + (size_t)b * Tc * (3 * Ec);

  v16h qa0, qa1;
  {
    const float* qrow = base + (size_t)(qb * 16 + r) * (3 * Ec) + h * Dhc + 16 * g;
    cvt4(qa0, 0,  *(const float4*)(qrow + 0));
    cvt4(qa0, 4,  *(const float4*)(qrow + 4));
    cvt4(qa0, 8,  *(const float4*)(qrow + 8));
    cvt4(qa0, 12, *(const float4*)(qrow + 12));
    cvt4(qa1, 0,  *(const float4*)(qrow + 32));
    cvt4(qa1, 4,  *(const float4*)(qrow + 36));
    cvt4(qa1, 8,  *(const float4*)(qrow + 40));
    cvt4(qa1, 12, *(const float4*)(qrow + 44));
  }
  float m_run[8], l_run[8], alpha[8];
  v8f o0 = {}, o1 = {}, o2 = {}, o3 = {};
#pragma unroll
  for (int i = 0; i < 8; ++i) { m_run[i] = -1e30f; l_run[i] = 0.f; }

  for (int kb = 0; kb < Tc; kb += 32) {
    v8f s0 = {}, s1 = {};
    {
      const float* k0row = base + (size_t)(kb + r)      * (3 * Ec) + Ec + h * Dhc + 16 * g;
      const float* k1row = base + (size_t)(kb + 16 + r) * (3 * Ec) + Ec + h * Dhc + 16 * g;
      v16h bb;
      cvt4(bb, 0, *(const float4*)(k0row + 0));  cvt4(bb, 4,  *(const float4*)(k0row + 4));
      cvt4(bb, 8, *(const float4*)(k0row + 8));  cvt4(bb, 12, *(const float4*)(k0row + 12));
      s0 = wmma_f16(qa0, bb, s0);
      cvt4(bb, 0, *(const float4*)(k0row + 32)); cvt4(bb, 4,  *(const float4*)(k0row + 36));
      cvt4(bb, 8, *(const float4*)(k0row + 40)); cvt4(bb, 12, *(const float4*)(k0row + 44));
      s0 = wmma_f16(qa1, bb, s0);
      cvt4(bb, 0, *(const float4*)(k1row + 0));  cvt4(bb, 4,  *(const float4*)(k1row + 4));
      cvt4(bb, 8, *(const float4*)(k1row + 8));  cvt4(bb, 12, *(const float4*)(k1row + 12));
      s1 = wmma_f16(qa0, bb, s1);
      cvt4(bb, 0, *(const float4*)(k1row + 32)); cvt4(bb, 4,  *(const float4*)(k1row + 36));
      cvt4(bb, 8, *(const float4*)(k1row + 40)); cvt4(bb, 12, *(const float4*)(k1row + 44));
      s1 = wmma_f16(qa1, bb, s1);
    }
    // online softmax per row (rows i+8*g), reductions across the 16-lane group
#pragma unroll
    for (int i = 0; i < 8; ++i) {
      float a0 = s0[i] * 0.125f, a1s = s1[i] * 0.125f;
      float mx = fmaxf(a0, a1s);
#pragma unroll
      for (int msk = 8; msk >= 1; msk >>= 1) mx = fmaxf(mx, __shfl_xor(mx, msk, 32));
      float mnew = fmaxf(m_run[i], mx);
      float al = __expf(m_run[i] - mnew);
      float p0 = __expf(a0 - mnew), p1 = __expf(a1s - mnew);
      float rs = p0 + p1;
#pragma unroll
      for (int msk = 8; msk >= 1; msk >>= 1) rs += __shfl_xor(rs, msk, 32);
      l_run[i] = l_run[i] * al + rs;
      m_run[i] = mnew;
      alpha[i] = al;
      s0[i] = p0; s1[i] = p1;
    }
#pragma unroll
    for (int i = 0; i < 8; ++i) { o0[i] *= alpha[i]; o1[i] *= alpha[i]; o2[i] *= alpha[i]; o3[i] *= alpha[i]; }

    // transpose P from C-layout to A-layout via LDS (1 KB, same wave)
#pragma unroll
    for (int i = 0; i < 8; ++i) {
      plds[i + 8*g][r]      = (_Float16)s0[i];
      plds[i + 8*g][16 + r] = (_Float16)s1[i];
    }
    __builtin_amdgcn_wave_barrier();
    v16h pa = cat8(*(const v8h*)(&plds[r][16 * g]), *(const v8h*)(&plds[r][8 + 16 * g]));
    __builtin_amdgcn_wave_barrier();

    const float* vbase = base + 2 * Ec + h * Dhc;
#pragma unroll
    for (int nt = 0; nt < 4; ++nt) {
      v16h vb;
#pragma unroll
      for (int j = 0; j < 8; ++j)
#pragma unroll
        for (int p = 0; p < 2; ++p) {
          int kk = 2*j + p + 16*g;
          vb[2*j+p] = (_Float16)vbase[(size_t)(kb + kk) * (3 * Ec) + nt * 16 + r];
        }
      if      (nt == 0) o0 = wmma_f16(pa, vb, o0);
      else if (nt == 1) o1 = wmma_f16(pa, vb, o1);
      else if (nt == 2) o2 = wmma_f16(pa, vb, o2);
      else              o3 = wmma_f16(pa, vb, o3);
    }
  }
#pragma unroll
  for (int i = 0; i < 8; ++i) {
    float inv = 1.f / l_run[i];
    int t = qb * 16 + i + 8*g;
    float* orow = att + ((size_t)(b * Tc + t)) * Ec + h * Dhc;
    orow[r]      = o0[i] * inv;
    orow[16 + r] = o1[i] * inv;
    orow[32 + r] = o2[i] * inv;
    orow[48 + r] = o3[i] * inv;
  }
}

// ================= router: noisy logits -> top2 gates =================
__global__ void router_topk_kernel(const float* __restrict__ logits, const float* __restrict__ nlog,
                                   const float* __restrict__ noise, float* __restrict__ gate,
                                   int* __restrict__ top2)
{
  int n = blockIdx.x * 256 + threadIdx.x;
  if (n >= BTc) return;
  float nv[Xc];
#pragma unroll
  for (int xi = 0; xi < Xc; ++xi) {
    float z = nlog[n * Xc + xi];
    float sp = (z > 20.f) ? z : log1pf(__expf(z));
    nv[xi] = logits[n * Xc + xi] + noise[n * Xc + xi] * sp;
  }
  int i1 = 0; float v1 = nv[0];
#pragma unroll
  for (int xi = 1; xi < Xc; ++xi) if (nv[xi] > v1) { v1 = nv[xi]; i1 = xi; }
  int i2 = -1; float v2 = -1e30f;
#pragma unroll
  for (int xi = 0; xi < Xc; ++xi) if (xi != i1 && nv[xi] > v2) { v2 = nv[xi]; i2 = xi; }
  float p1 = 1.f / (1.f + __expf(v2 - v1));
  float p2 = 1.f - p1;
#pragma unroll
  for (int xi = 0; xi < Xc; ++xi) gate[n * Xc + xi] = 0.f;
  gate[n * Xc + i1] = p1;
  gate[n * Xc + i2] = p2;
  top2[2 * n] = i1; top2[2 * n + 1] = i2;
}

// ======= per-expert token lists, deterministic block scan (token order) =======
__global__ __launch_bounds__(256)
void build_lists_kernel(const int* __restrict__ top2, int* __restrict__ idxlist, int* __restrict__ counts)
{
  __shared__ int sc[256];
  __shared__ int sbase;
  const int e = blockIdx.x, tid = threadIdx.x;
  int* lst = idxlist + e * CAPc;
  if (tid == 0) sbase = 0;
  __syncthreads();
  for (int chunk = 0; chunk < BTc; chunk += 256) {
    int n = chunk + tid;
    int flag = (top2[2 * n] == e || top2[2 * n + 1] == e) ? 1 : 0;
    sc[tid] = flag;
    __syncthreads();
    for (int off = 1; off < 256; off <<= 1) {
      int v = (tid >= off) ? sc[tid - off] : 0;
      __syncthreads();
      sc[tid] += v;
      __syncthreads();
    }
    int slot = sbase + sc[tid] - 1;
    if (flag && slot < CAPc) lst[slot] = n;
    int total = sc[255];
    __syncthreads();
    if (tid == 0) sbase += total;
    __syncthreads();
  }
  if (tid == 0) counts[e] = (sbase > CAPc) ? CAPc : sbase;
}

__global__ void gather_kernel(const float* __restrict__ h, const int* __restrict__ lst,
                              const int* __restrict__ count, float* __restrict__ tok)
{
  int id = blockIdx.x * 256 + threadIdx.x;    // CAP*E
  int slot = id >> 9, e = id & (Ec - 1);
  int c = *count;
  tok[id] = (slot < c) ? h[(size_t)lst[slot] * Ec + e] : 0.f;
}

__global__ void scatter_kernel(const float* __restrict__ eo, const int* __restrict__ lst,
                               const int* __restrict__ count, const float* __restrict__ gate,
                               int expert, float* __restrict__ x)
{
  int id = blockIdx.x * 256 + threadIdx.x;    // CAP*E
  int slot = id >> 9, e = id & (Ec - 1);
  if (slot < *count) {
    int n = lst[slot];
    x[(size_t)n * Ec + e] += eo[id] * gate[n * Xc + expert];
  }
}

// ================= pooling + head =================
__global__ void pool_score_kernel(const float* __restrict__ xf, const float* __restrict__ pq,
                                  float* __restrict__ s)
{
  int n = blockIdx.x * 256 + threadIdx.x;     // BT
  float acc = 0.f;
  for (int e = 0; e < Ec; ++e) acc += pq[e] * xf[(size_t)n * Ec + e];
  s[n] = acc * 0.044194173824159216f;         // 1/sqrt(512)
}

__global__ __launch_bounds__(256)
void pool_softmax_kernel(const float* __restrict__ s, float* __restrict__ w)
{
  __shared__ float red[256];
  const int b = blockIdx.x, tid = threadIdx.x;
  const float* sb = s + b * Tc;
  float v[4]; float mx = -1e30f;
#pragma unroll
  for (int k = 0; k < 4; ++k) { v[k] = sb[tid + 256 * k]; mx = fmaxf(mx, v[k]); }
  red[tid] = mx; __syncthreads();
  for (int off = 128; off > 0; off >>= 1) { if (tid < off) red[tid] = fmaxf(red[tid], red[tid + off]); __syncthreads(); }
  float m = red[0];
  __syncthreads();
  float sum = 0.f;
#pragma unroll
  for (int k = 0; k < 4; ++k) { v[k] = __expf(v[k] - m); sum += v[k]; }
  red[tid] = sum; __syncthreads();
  for (int off = 128; off > 0; off >>= 1) { if (tid < off) red[tid] += red[tid + off]; __syncthreads(); }
  float inv = 1.f / red[0];
#pragma unroll
  for (int k = 0; k < 4; ++k) w[b * Tc + tid + 256 * k] = v[k] * inv;
}

__global__ void pool_reduce_kernel(const float* __restrict__ xf, const float* __restrict__ w,
                                   float* __restrict__ pooled)
{
  int id = blockIdx.x * 256 + threadIdx.x;    // B*E
  int b = id >> 9, e = id & (Ec - 1);
  float acc = 0.f;
  for (int t = 0; t < Tc; ++t) acc += w[b * Tc + t] * xf[((size_t)b * Tc + t) * Ec + e];
  pooled[id] = acc;
}

__global__ __launch_bounds__(256)
void head2_kernel(const float* __restrict__ h1, const float* __restrict__ w2,
                  const float* __restrict__ b2, float* __restrict__ out)
{
  __shared__ float red[256];
  const int b = blockIdx.x, tid = threadIdx.x;
  float s = 0.f;
#pragma unroll
  for (int k = 0; k < 4; ++k) { int e = tid + 256 * k; s += h1[b * HIDc + e] * w2[e]; }
  red[tid] = s; __syncthreads();
  for (int off = 128; off > 0; off >>= 1) { if (tid < off) red[tid] += red[tid + off]; __syncthreads(); }
  if (tid == 0) out[b] = red[0] + b2[0];
}

// ================= driver =================
extern "C" void kernel_launch(void* const* d_in, const int* in_sizes, int n_in,
                              void* d_out, int out_size, void* d_ws, size_t ws_size,
                              hipStream_t stream)
{
  (void)in_sizes; (void)n_in; (void)out_size; (void)ws_size;
  const float* tok_emb = (const float*)d_in[0];
  const float* pos_emb = (const float*)d_in[1];
  const float* ln1_w   = (const float*)d_in[2];
  const float* ln1_b   = (const float*)d_in[3];
  const float* ln2_w   = (const float*)d_in[4];
  const float* ln2_b   = (const float*)d_in[5];
  const float* qkv_w   = (const float*)d_in[6];
  const float* out_w   = (const float*)d_in[7];
  const float* route_w = (const float*)d_in[8];
  const float* route_b = (const float*)d_in[9];
  const float* noise_w = (const float*)d_in[10];
  const float* noise_b = (const float*)d_in[11];
  const float* exp_w1  = (const float*)d_in[12];
  const float* exp_b1  = (const float*)d_in[13];
  const float* exp_w2  = (const float*)d_in[14];
  const float* exp_b2  = (const float*)d_in[15];
  const float* lnf_w   = (const float*)d_in[16];
  const float* lnf_b   = (const float*)d_in[17];
  const float* pool_q  = (const float*)d_in[18];
  const float* pool_w  = (const float*)d_in[19];
  const float* pool_b  = (const float*)d_in[20];
  const float* head_w1 = (const float*)d_in[21];
  const float* head_b1 = (const float*)d_in[22];
  const float* head_w2 = (const float*)d_in[23];
  const float* head_b2 = (const float*)d_in[24];
  const int*   ids     = (const int*)d_in[25];
  const float* rnoise  = (const float*)d_in[26];

  char* wp = (char*)d_ws;
  auto alloc = [&](size_t bytes) -> void* {
    void* p = wp; wp += ((bytes + 255) / 256) * 256; return p;
  };
  float* x       = (float*)alloc((size_t)BTc * Ec * 4);
  float* hbuf    = (float*)alloc((size_t)BTc * Ec * 4);
  float* qkvb    = (float*)alloc((size_t)BTc * 3 * Ec * 4);
  float* attb    = (float*)alloc((size_t)BTc * Ec * 4);
  float* logitsb = (float*)alloc((size_t)BTc * Xc * 4);
  float* nlogb   = (float*)alloc((size_t)BTc * Xc * 4);
  float* gateb   = (float*)alloc((size_t)BTc * Xc * 4);
  int*   top2b   = (int*)  alloc((size_t)BTc * 2 * 4);
  int*   idxl    = (int*)  alloc((size_t)Xc * CAPc * 4);
  int*   cnts    = (int*)  alloc((size_t)Xc * 4);
  float* tokb    = (float*)alloc((size_t)CAPc * Ec * 4);
  float* actb    = (float*)alloc((size_t)CAPc * 4 * Ec * 4);
  float* eob     = (float*)alloc((size_t)CAPc * Ec * 4);
  float* sbuf    = (float*)alloc((size_t)BTc * 4);
  float* wbuf    = (float*)alloc((size_t)BTc * 4);
  float* pooledb = (float*)alloc((size_t)Bc * Ec * 4);
  float* hvb     = (float*)alloc((size_t)Bc * HIGHc * 4);
  float* h1b     = (float*)alloc((size_t)Bc * HIDc * 4);

  auto gemm = [&](const float* A, const float* W, const float* bias, float* C,
                  int M, int N, int Kd, int flags) {
    dim3 grid((N + 127) / 128, (M + 63) / 64);
    gemm_nt_kernel<<<grid, 256, 0, stream>>>(A, W, bias, C, M, N, Kd, flags);
  };

  embed_kernel<<<(BTc * Ec) / 256, 256, 0, stream>>>(tok_emb, pos_emb, ids, x);

  for (int l = 0; l < Lc; ++l) {
    layernorm_kernel<<<BTc, 256, 0, stream>>>(x, ln1_w + l * Ec, ln1_b + l * Ec, hbuf);
    gemm(hbuf, qkv_w + (size_t)l * 3 * Ec * Ec, nullptr, qkvb, BTc, 3 * Ec, Ec, 0);
    rope_kernel<<<(BTc * Hc * 32 + 255) / 256, 256, 0, stream>>>(qkvb);
    attn_kernel<<<dim3(Tc / 16, Hc, Bc), 32, 0, stream>>>(qkvb, attb);
    gemm(attb, out_w + (size_t)l * Ec * Ec, nullptr, x, BTc, Ec, Ec, 4 /*accumulate*/);

    layernorm_kernel<<<BTc, 256, 0, stream>>>(x, ln2_w + l * Ec, ln2_b + l * Ec, hbuf);
    gemm(hbuf, route_w + (size_t)l * Xc * Ec, route_b + l * Xc, logitsb, BTc, Xc, Ec, 1);
    gemm(hbuf, noise_w + (size_t)l * Xc * Ec, noise_b + l * Xc, nlogb, BTc, Xc, Ec, 1);
    router_topk_kernel<<<BTc / 256, 256, 0, stream>>>(logitsb, nlogb,
                                                      rnoise + (size_t)l * BTc * Xc, gateb, top2b);
    build_lists_kernel<<<Xc, 256, 0, stream>>>(top2b, idxl, cnts);

    for (int i = 0; i < Xc; ++i) {
      gather_kernel<<<(CAPc * Ec) / 256, 256, 0, stream>>>(hbuf, idxl + i * CAPc, cnts + i, tokb);
      gemm(tokb, exp_w1 + (size_t)(l * Xc + i) * 4 * Ec * Ec,
           exp_b1 + (size_t)(l * Xc + i) * 4 * Ec, actb, CAPc, 4 * Ec, Ec, 1 | 2);
      gemm(actb, exp_w2 + (size_t)(l * Xc + i) * Ec * 4 * Ec,
           exp_b2 + (size_t)(l * Xc + i) * Ec, eob, CAPc, Ec, 4 * Ec, 1);
      scatter_kernel<<<(CAPc * Ec) / 256, 256, 0, stream>>>(eob, idxl + i * CAPc, cnts + i,
                                                            gateb, i, x);
    }
  }

  layernorm_kernel<<<BTc, 256, 0, stream>>>(x, lnf_w, lnf_b, hbuf);
  pool_score_kernel<<<BTc / 256, 256, 0, stream>>>(hbuf, pool_q, sbuf);
  pool_softmax_kernel<<<Bc, 256, 0, stream>>>(sbuf, wbuf);
  pool_reduce_kernel<<<(Bc * Ec) / 256, 256, 0, stream>>>(hbuf, wbuf, pooledb);
  gemm(pooledb, pool_w, pool_b, hvb, Bc, HIGHc, Ec, 1);
  gemm(hvb, head_w1, head_b1, h1b, Bc, HIDc, HIGHc, 1 | 2);
  head2_kernel<<<Bc, 256, 0, stream>>>(h1b, head_w2, head_b2, (float*)d_out);
}
